// MultiheadAttention_26027501814394
// MI455X (gfx1250) — compile-verified, matrix operand order checked by reading, not run on hardware
//
#include <hip/hip_runtime.h>
#include <hip/hip_bf16.h>

typedef __bf16 bf16;
typedef __attribute__((ext_vector_type(16))) __bf16 v16bf;
typedef __attribute__((ext_vector_type(8)))  __bf16 bf16x8;
typedef __attribute__((ext_vector_type(4)))  __bf16 bf16x4;
typedef __attribute__((ext_vector_type(8)))  float  v8f;
typedef int v4i_vs __attribute__((vector_size(16)));  // matches builtin's param

#if __has_builtin(__builtin_amdgcn_global_load_async_to_lds_b128) && \
    __has_builtin(__builtin_amdgcn_s_wait_asynccnt)
#define HAVE_ASYNC_LDS 1
#else
#define HAVE_ASYNC_LDS 0
#endif

__device__ __forceinline__ void async_copy_b128(const bf16* g, bf16* l) {
#if HAVE_ASYNC_LDS
    __builtin_amdgcn_global_load_async_to_lds_b128(
        (__attribute__((address_space(1))) v4i_vs*)(uintptr_t)g,
        (__attribute__((address_space(3))) v4i_vs*)(unsigned)(uintptr_t)l,
        0, 0);
#endif
}
__device__ __forceinline__ void wait_async0() {
#if HAVE_ASYNC_LDS
    __builtin_amdgcn_s_wait_asynccnt(0);
#endif
}

__device__ __forceinline__ v8f vzero8() {
    v8f z; for (int i = 0; i < 8; ++i) z[i] = 0.0f; return z;
}

union FragU { v16bf v; bf16x8 h8[2]; };

__device__ __forceinline__ bf16x4 cvt4(float4 d) {
    bf16x4 t;
    t[0] = (bf16)d.x; t[1] = (bf16)d.y; t[2] = (bf16)d.z; t[3] = (bf16)d.w;
    return t;
}

// ---------------------------------------------------------------------------
// GEMM: C[M,N] = A[M,K] @ W[N,K]^T   (torch Linear convention)
// A is fp32 (ABF16=false) or bf16 (ABF16=true). W is fp32.
// OUTMODE 0: C fp32 row-major [M,N]
// OUTMODE 1: C bf16 scattered to [B, heads, S, D] with n = h*128 + d, m = b*S+s
// 256 threads (8 waves), 128x128 tile, K-step 32, double-buffered LDS.
// ---------------------------------------------------------------------------
template <bool ABF16, int OUTMODE>
__global__ __launch_bounds__(256) void gemm_wmma(
    const void* __restrict__ Aptr, const float* __restrict__ W,
    void* __restrict__ Cptr, int M, int N, int K, int heads, int S) {
    constexpr int LDSTR = 40;  // 32 + 8 pad, 80B rows (16B aligned)
    __shared__ __align__(16) bf16 As[2][128][LDSTR];
    __shared__ __align__(16) bf16 Bs[2][128][LDSTR];

    const int tid  = threadIdx.x;
    const int lane = tid & 31;
    const int wave = tid >> 5;
    const int bm = blockIdx.y * 128;
    const int bn = blockIdx.x * 128;
    const int wm = (wave >> 1) * 32;   // 0,32,64,96
    const int wn = (wave & 1) * 64;    // 0,64
    const int lr = lane & 15;
    const int kb = (lane >> 4) * 8;    // 0 or 8 (lane-half K base)

    // staging coordinates (fp32 path: 4 x float4; bf16 path: 2 x bf16x8)
    const int fr[4] = { (tid + 0) >> 3, (tid + 256) >> 3,
                        (tid + 512) >> 3, (tid + 768) >> 3 };
    const int fcc   = (tid & 7) * 4;
    const int hr[2] = { (tid + 0) >> 2, (tid + 256) >> 2 };
    const int hc    = (tid & 3) * 8;

    v8f acc[2][4];
#pragma unroll
    for (int i = 0; i < 2; ++i)
#pragma unroll
        for (int j = 0; j < 4; ++j) acc[i][j] = vzero8();

    // ---- prologue: stage K-step 0 into buffer 0 ----
    if (ABF16) {
        const bf16* A = (const bf16*)Aptr;
#pragma unroll
        for (int i = 0; i < 2; ++i)
            *(bf16x8*)&As[0][hr[i]][hc] =
                *(const bf16x8*)&A[(size_t)(bm + hr[i]) * K + hc];
    } else {
        const float* A = (const float*)Aptr;
#pragma unroll
        for (int i = 0; i < 4; ++i)
            *(bf16x4*)&As[0][fr[i]][fcc] =
                cvt4(*(const float4*)&A[(size_t)(bm + fr[i]) * K + fcc]);
    }
#pragma unroll
    for (int i = 0; i < 4; ++i)
        *(bf16x4*)&Bs[0][fr[i]][fcc] =
            cvt4(*(const float4*)&W[(size_t)(bn + fr[i]) * K + fcc]);

    int cur = 0;
    for (int k0 = 0; k0 < K; k0 += 32) {
        const bool have_next = (k0 + 32) < K;
        // ---- issue next tile's global loads into registers ----
        float4 ra[4], rb[4];
        bf16x8 rh[2];
        if (have_next) {
            const int kn = k0 + 32;
            if (ABF16) {
                const bf16* A = (const bf16*)Aptr;
#pragma unroll
                for (int i = 0; i < 2; ++i)
                    rh[i] = *(const bf16x8*)&A[(size_t)(bm + hr[i]) * K + kn + hc];
            } else {
                const float* A = (const float*)Aptr;
#pragma unroll
                for (int i = 0; i < 4; ++i)
                    ra[i] = *(const float4*)&A[(size_t)(bm + fr[i]) * K + kn + fcc];
            }
#pragma unroll
            for (int i = 0; i < 4; ++i)
                rb[i] = *(const float4*)&W[(size_t)(bn + fr[i]) * K + kn + fcc];
        }

        __syncthreads();  // buf[cur] fully staged

        // ---- fragments + 8 WMMAs on buf[cur] ----
        FragU af[2], bfr[4];
#pragma unroll
        for (int i = 0; i < 2; ++i) {
            const bf16* p = &As[cur][wm + i * 16 + lr][kb];
            af[i].h8[0] = *(const bf16x8*)p;
            af[i].h8[1] = *(const bf16x8*)(p + 16);
        }
#pragma unroll
        for (int j = 0; j < 4; ++j) {
            const bf16* p = &Bs[cur][wn + j * 16 + lr][kb];
            bfr[j].h8[0] = *(const bf16x8*)p;
            bfr[j].h8[1] = *(const bf16x8*)(p + 16);
        }
#pragma unroll
        for (int i = 0; i < 2; ++i)
#pragma unroll
            for (int j = 0; j < 4; ++j)
                acc[i][j] = __builtin_amdgcn_wmma_f32_16x16x32_bf16(
                    false, af[i].v, false, bfr[j].v, (short)0, acc[i][j],
                    false, false);

        // ---- store next tile into the alternate buffer ----
        if (have_next) {
            const int nxt = cur ^ 1;
            if (ABF16) {
#pragma unroll
                for (int i = 0; i < 2; ++i)
                    *(bf16x8*)&As[nxt][hr[i]][hc] = rh[i];
            } else {
#pragma unroll
                for (int i = 0; i < 4; ++i)
                    *(bf16x4*)&As[nxt][fr[i]][fcc] = cvt4(ra[i]);
            }
#pragma unroll
            for (int i = 0; i < 4; ++i)
                *(bf16x4*)&Bs[nxt][fr[i]][fcc] = cvt4(rb[i]);
        }
        cur ^= 1;
    }

    // ---- epilogue: C layout = lane holds col (lane&15), rows v + 8*(lane>>4)
    const int hb = (lane >> 4) * 8;
#pragma unroll
    for (int i = 0; i < 2; ++i)
#pragma unroll
        for (int j = 0; j < 4; ++j)
#pragma unroll
            for (int v = 0; v < 8; ++v) {
                int m = bm + wm + i * 16 + v + hb;
                int n = bn + wn + j * 16 + lr;
                if (OUTMODE == 0) {
                    ((float*)Cptr)[(size_t)m * N + n] = acc[i][j][v];
                } else {
                    int bb = m / S, s = m % S;
                    int hh = n >> 7, dd = n & 127;
                    ((bf16*)Cptr)[(((size_t)(bb * heads + hh)) * S + s) * 128 +
                                  dd] = (bf16)acc[i][j][v];
                }
            }
}

// ---------------------------------------------------------------------------
// Flash attention (causal, GQA rep=4). Block = 4 waves, 64 query rows.
// Q bf16 [B,16,S,128], K/V bf16 [B,4,S,128], out bf16 [B,S,16,128].
// K staged via async global->LDS DMA (ASYNCcnt) when the toolchain has it;
// V staged via registers with on-the-fly transpose. Double-buffered.
// ---------------------------------------------------------------------------
__global__ __launch_bounds__(128) void flash_attn(
    const bf16* __restrict__ Q, const bf16* __restrict__ Kg,
    const bf16* __restrict__ Vg, bf16* __restrict__ O, int S) {
    constexpr int D = 128;
    __shared__ __align__(16) bf16 Ks[2][32][136];   // keys row-major
    __shared__ __align__(16) bf16 Vt[2][128][40];   // V transposed [d][t]
    __shared__ __align__(16) bf16 Ps[4][16][40];    // per-wave P tile

    const int tid  = threadIdx.x;
    const int lane = tid & 31;
    const int wave = tid >> 5;
    const int qb = blockIdx.x * 64;
    const int h  = blockIdx.y;
    const int b  = blockIdx.z;
    const int kvh = h >> 2;                      // 16 q-heads / 4 kv-heads
    const int lr = lane & 15;
    const int hb = (lane >> 4) * 8;
    const int qrow0 = qb + wave * 16;
    const int wq_max_s = __builtin_amdgcn_readfirstlane(qrow0 + 15);

    const bf16* qbase = Q + ((size_t)b * 16 + h) * S * D;
    const bf16* kbase = Kg + ((size_t)b * 4 + kvh) * S * D;
    const bf16* vbase = Vg + ((size_t)b * 4 + kvh) * S * D;

    // staging coordinates: 512 chunks of 8 bf16 over 128 threads
    const int sr[4] = { (tid + 0) >> 4, (tid + 128) >> 4,
                        (tid + 256) >> 4, (tid + 384) >> 4 };
    const int sc_ = (tid & 15) * 8;

    // Q fragments live in registers for the whole kernel (16 x 128 = 4 frags)
    FragU qf[4];
    {
        const bf16* qp = qbase + (size_t)(qrow0 + lr) * D;
#pragma unroll
        for (int kk = 0; kk < 4; ++kk) {
            qf[kk].h8[0] = *(const bf16x8*)(qp + kk * 32 + hb);
            qf[kk].h8[1] = *(const bf16x8*)(qp + kk * 32 + hb + 16);
        }
    }

    v8f o[8];
#pragma unroll
    for (int j = 0; j < 8; ++j) o[j] = vzero8();
    float mrow[8], lrow[8];
#pragma unroll
    for (int v = 0; v < 8; ++v) { mrow[v] = -1e30f; lrow[v] = 0.0f; }
    const float sc = 0.08838834764831845f;  // 1/sqrt(128)
    const int tmax = qb + 64;

    // ---- prologue: stage t-block 0 into buffer 0 ----
#if HAVE_ASYNC_LDS
#pragma unroll
    for (int i = 0; i < 4; ++i)
        async_copy_b128(&kbase[(size_t)sr[i] * D + sc_], &Ks[0][sr[i]][sc_]);
#else
#pragma unroll
    for (int i = 0; i < 4; ++i)
        *(bf16x8*)&Ks[0][sr[i]][sc_] =
            *(const bf16x8*)&kbase[(size_t)sr[i] * D + sc_];
#endif
#pragma unroll
    for (int i = 0; i < 4; ++i) {
        bf16x8 d = *(const bf16x8*)&vbase[(size_t)sr[i] * D + sc_];
#pragma unroll
        for (int e = 0; e < 8; ++e) Vt[0][sc_ + e][sr[i]] = d[e];
    }

    int cur = 0;
    for (int t0 = 0; t0 < tmax; t0 += 32) {
        const bool have_next = (t0 + 32) < tmax;
        const int tn = t0 + 32;
        // ---- issue next V block's global loads (registers) ----
        bf16x8 rv[4];
#if !HAVE_ASYNC_LDS
        bf16x8 rk[4];
#endif
        if (have_next) {
#pragma unroll
            for (int i = 0; i < 4; ++i) {
                rv[i] = *(const bf16x8*)&vbase[(size_t)(tn + sr[i]) * D + sc_];
#if !HAVE_ASYNC_LDS
                rk[i] = *(const bf16x8*)&kbase[(size_t)(tn + sr[i]) * D + sc_];
#endif
            }
        }

        wait_async0();    // my K-DMA into buf[cur] has landed
        __syncthreads();  // everyone's staging of buf[cur] is visible

#if HAVE_ASYNC_LDS
        // issue next K block's DMA now: overlaps with the whole compute phase
        if (have_next) {
            const int nxt = cur ^ 1;
#pragma unroll
            for (int i = 0; i < 4; ++i)
                async_copy_b128(&kbase[(size_t)(tn + sr[i]) * D + sc_],
                                &Ks[nxt][sr[i]][sc_]);
        }
#endif

        if (t0 <= wq_max_s) {  // scalar branch: EXEC stays all-ones for WMMA
            // ---- scores: Q (16x128) @ K^T (128x32) -> two 16x16 C tiles
            v8f s0 = vzero8(), s1 = vzero8();
#pragma unroll
            for (int kk = 0; kk < 4; ++kk) {
                FragU kf;
                kf.h8[0] = *(const bf16x8*)&Ks[cur][lr][kk * 32 + hb];
                kf.h8[1] = *(const bf16x8*)&Ks[cur][lr][kk * 32 + hb + 16];
                s0 = __builtin_amdgcn_wmma_f32_16x16x32_bf16(
                    false, qf[kk].v, false, kf.v, (short)0, s0, false, false);
                kf.h8[0] = *(const bf16x8*)&Ks[cur][16 + lr][kk * 32 + hb];
                kf.h8[1] = *(const bf16x8*)&Ks[cur][16 + lr][kk * 32 + hb + 16];
                s1 = __builtin_amdgcn_wmma_f32_16x16x32_bf16(
                    false, qf[kk].v, false, kf.v, (short)0, s1, false, false);
            }
            // ---- online softmax in C-register layout ----
            const int col0 = t0 + lr, col1 = t0 + 16 + lr;
#pragma unroll
            for (int v = 0; v < 8; ++v) {
                int row = qrow0 + v + hb;
                float a = s0[v] * sc, c2 = s1[v] * sc;
                if (col0 > row) a = -1e30f;
                if (col1 > row) c2 = -1e30f;
                float mx = fmaxf(a, c2);
#pragma unroll
                for (int m = 8; m >= 1; m >>= 1)
                    mx = fmaxf(mx, __shfl_xor(mx, m, 32));
                float mnew = fmaxf(mrow[v], mx);
                float p0 = __expf(a - mnew);
                float p1 = __expf(c2 - mnew);
                float scale = __expf(mrow[v] - mnew);
                float rs = p0 + p1;
#pragma unroll
                for (int m = 8; m >= 1; m >>= 1) rs += __shfl_xor(rs, m, 32);
                lrow[v] = lrow[v] * scale + rs;
                mrow[v] = mnew;
#pragma unroll
                for (int j = 0; j < 8; ++j) o[j][v] *= scale;
                // C-layout -> LDS for A-layout reload
                Ps[wave][v + hb][lr]      = (bf16)p0;
                Ps[wave][v + hb][lr + 16] = (bf16)p1;
            }
            // ---- PV: P (16x32) @ V (32x128) -> 8 accum tiles ----
            FragU pf;
            pf.h8[0] = *(const bf16x8*)&Ps[wave][lr][hb];
            pf.h8[1] = *(const bf16x8*)&Ps[wave][lr][hb + 16];
#pragma unroll
            for (int j = 0; j < 8; ++j) {
                FragU vf;
                vf.h8[0] = *(const bf16x8*)&Vt[cur][j * 16 + lr][hb];
                vf.h8[1] = *(const bf16x8*)&Vt[cur][j * 16 + lr][hb + 16];
                o[j] = __builtin_amdgcn_wmma_f32_16x16x32_bf16(
                    false, pf.v, false, vf.v, (short)0, o[j], false, false);
            }
        }

        // ---- store next block into alternate buffers ----
        if (have_next) {
            const int nxt = cur ^ 1;
#if !HAVE_ASYNC_LDS
#pragma unroll
            for (int i = 0; i < 4; ++i)
                *(bf16x8*)&Ks[nxt][sr[i]][sc_] = rk[i];
#endif
#pragma unroll
            for (int i = 0; i < 4; ++i)
#pragma unroll
                for (int e = 0; e < 8; ++e) Vt[nxt][sc_ + e][sr[i]] = rv[i][e];
        }
        cur ^= 1;
    }

    // ---- epilogue: out[b, s, h, d] bf16 ----
#pragma unroll
    for (int v = 0; v < 8; ++v) {
        float inv = 1.0f / lrow[v];
        int s = qrow0 + v + hb;
        bf16* op = O + (((size_t)b * S + s) * 16 + h) * D;
#pragma unroll
        for (int j = 0; j < 8; ++j)
            op[j * 16 + lr] = (bf16)(o[j][v] * inv);
    }
}

// ---------------------------------------------------------------------------
extern "C" void kernel_launch(void* const* d_in, const int* in_sizes, int n_in,
                              void* d_out, int out_size, void* d_ws,
                              size_t ws_size, hipStream_t stream) {
    (void)in_sizes; (void)n_in; (void)out_size; (void)ws_size;
    const float* x  = (const float*)d_in[0];
    const float* Wq = (const float*)d_in[1];
    const float* Wk = (const float*)d_in[2];
    const float* Wv = (const float*)d_in[3];
    const float* Wo = (const float*)d_in[4];

    const int B = 2, S = 2048, E = 2048, H = 16, KV = 4, D = 128;
    const int M = B * S;

    bf16* qbuf = (bf16*)d_ws;                      // [B,H,S,D]
    bf16* kbuf = qbuf + (size_t)B * H * S * D;     // [B,KV,S,D]
    bf16* vbuf = kbuf + (size_t)B * KV * S * D;    // [B,KV,S,D]
    bf16* abuf = vbuf + (size_t)B * KV * S * D;    // [B,S,H,D]

    dim3 blk(256);
    gemm_wmma<false, 1><<<dim3(E / 128, M / 128), blk, 0, stream>>>(
        x, Wq, qbuf, M, E, E, H, S);
    gemm_wmma<false, 1><<<dim3((KV * D) / 128, M / 128), blk, 0, stream>>>(
        x, Wk, kbuf, M, KV * D, E, KV, S);
    gemm_wmma<false, 1><<<dim3((KV * D) / 128, M / 128), blk, 0, stream>>>(
        x, Wv, vbuf, M, KV * D, E, KV, S);

    flash_attn<<<dim3(S / 64, H, B), dim3(128), 0, stream>>>(
        qbuf, kbuf, vbuf, abuf, S);

    gemm_wmma<true, 0><<<dim3(E / 128, M / 128), blk, 0, stream>>>(
        abuf, Wo, d_out, M, E, E, 0, S);
}
